// TSModel_45165876084961
// MI455X (gfx1250) — matrix operand order repaired and checked by reading, not compiled, not run on hardware
//
#include <hip/hip_runtime.h>

// ---------------------------------------------------------------------------
// Fused  l2norm(time) -> LSTM(64) -> LSTM(1) -> dense(1)   for MI455X/gfx1250
//   x : [256, 2048, 1]  fp32      out : [256, 2048, 1] fp32
// One workgroup (8 wave32s, 256 threads) owns 16 batch rows for the entire
// 2048-step recurrence. Recurrent weights live in registers as f16 WMMA B
// fragments; h-state lives in LDS (f16); c-state lives in registers.
// Activations use branch-free native v_exp_f32 / v_rcp_f32 TRANS ops so the
// serial per-step critical path is straight-line code.
// ---------------------------------------------------------------------------

typedef _Float16 v16h __attribute__((ext_vector_type(16)));
typedef _Float16 v8h  __attribute__((ext_vector_type(8)));
typedef float    v8f  __attribute__((ext_vector_type(8)));

#define TSTEPS 2048
#define UNITS  64
#define GATES  256   // 4 * UNITS
#define BPW    16    // batch rows per workgroup

#define LOG2E 1.44269504088896340736f

// sigmoid(x) = 1 / (1 + 2^(-x*log2e));  exp2 overflow/underflow saturate
// correctly through v_rcp (inf -> 0, 0 -> 1). Branch-free: 1 TRANS + FMA + RCP.
__device__ __forceinline__ float sigf(float v) {
    return __builtin_amdgcn_rcpf(1.0f + __builtin_amdgcn_exp2f(-LOG2E * v));
}

// tanh(x) = 1 - 2/(2^(2x*log2e) + 1); saturates exactly to +-1 at the ends.
__device__ __forceinline__ float tanhfast(float v) {
    return 1.0f - 2.0f * __builtin_amdgcn_rcpf(
               __builtin_amdgcn_exp2f((2.0f * LOG2E) * v) + 1.0f);
}

__launch_bounds__(256, 1)
__global__ void lstm_fused_kernel(const float* __restrict__ x,
                                  const float* __restrict__ W0,   // (65, 256)
                                  const float* __restrict__ b0,   // (256)
                                  const float* __restrict__ W1,   // (65, 4)
                                  const float* __restrict__ b1,   // (4)
                                  const float* __restrict__ Wd,   // (1,1)
                                  const float* __restrict__ bd,   // (1)
                                  float* __restrict__ out)        // (256, 2048)
{
    __shared__ float zbuf[BPW * GATES];                    // 16 KB gate staging
    __shared__ __align__(16) _Float16 h0[BPW * UNITS];     // layer-0 h state (f16)
    __shared__ float b0s[GATES];
    __shared__ float w0r0[GATES];                          // W0 row 0 (input weight)
    __shared__ float W1s[65 * 4];
    __shared__ float xbuf[BPW];
    __shared__ float invn[BPW];
    __shared__ float red[256];

    const int tid   = threadIdx.x;
    const int wave  = tid >> 5;
    const int lane  = tid & 31;
    const int hi    = lane >> 4;       // half-wave index (WMMA layout)
    const int ln    = lane & 15;
    const int bbase = blockIdx.x * BPW;

    // ---- preload constants into LDS, zero h state ----
    if (tid < GATES) { b0s[tid] = b0[tid]; w0r0[tid] = W0[tid]; }
    for (int i = tid; i < 65 * 4; i += 256) W1s[i] = W1[i];
    for (int i = tid; i < BPW * UNITS; i += 256) h0[i] = (_Float16)0.0f;

    // ---- phase 1: per-batch sum of squares over time -> 1/||x_b|| ----
    {
        const int r = tid >> 4, s = tid & 15;
        const float* xr = x + (size_t)(bbase + r) * TSTEPS;
        float acc = 0.0f;
        for (int k = s; k < TSTEPS; k += 16) { float v = xr[k]; acc += v * v; }
        red[tid] = acc;
        __syncthreads();
        if (tid < BPW) {
            float tot = 0.0f;
            for (int s2 = 0; s2 < 16; ++s2) tot += red[tid * 16 + s2];
            invn[tid] = __builtin_amdgcn_rsqf(fmaxf(tot, 1e-12f));
        }
    }

    // ---- phase 2: preload recurrent weights as f16 WMMA B fragments ----
    // Wave w owns N-tiles 2w, 2w+1.  B (16-bit, 32x16) layout:
    //   lane l: n = l&15 ; half j -> k = 32*kh + (j<8 ? 8*(l>>4)+j : 16+8*(l>>4)+j-8)
    v16h Bfrag[2][2];
    for (int tt = 0; tt < 2; ++tt) {
        const int n = (2 * wave + tt) * 16 + ln;
        for (int kh = 0; kh < 2; ++kh) {
            v16h bf;
            for (int j = 0; j < 16; ++j) {
                const int k = 32 * kh + (j < 8 ? 8 * hi + j : 16 + 8 * hi + (j - 8));
                bf[j] = (_Float16)W0[(size_t)(1 + k) * GATES + n];
            }
            Bfrag[tt][kh] = bf;
        }
    }

    // layer-0 cell state: thread owns cells idx = tid + 256*r2 (m=idx>>6,u=idx&63)
    float creg[4] = {0.f, 0.f, 0.f, 0.f};

    // layer-1 (units=1) + dense state: threads 0..15 own one batch row each
    float c1 = 0.f, h1 = 0.f;
    const float b1v0 = b1[0], b1v1 = b1[1], b1v2 = b1[2], b1v3 = b1[3];
    const float wdv = Wd[0], bdv = bd[0];

    auto layer1_step = [&](int tstore) {
        const int m = tid;
        float z0 = b1v0, zj1 = b1v1, z2 = b1v2, z3 = b1v3;
        #pragma unroll 8
        for (int u = 0; u < UNITS; ++u) {
            const float hv = (float)h0[m * UNITS + u];
            z0  += hv * W1s[u * 4 + 0];
            zj1 += hv * W1s[u * 4 + 1];
            z2  += hv * W1s[u * 4 + 2];
            z3  += hv * W1s[u * 4 + 3];
        }
        z0  += h1 * W1s[64 * 4 + 0];
        zj1 += h1 * W1s[64 * 4 + 1];
        z2  += h1 * W1s[64 * 4 + 2];
        z3  += h1 * W1s[64 * 4 + 3];
        c1 = sigf(z2 + 1.0f) * c1 + sigf(z0) * tanhfast(zj1);
        h1 = sigf(z3) * tanhfast(c1);
        out[(size_t)(bbase + m) * TSTEPS + tstore] = h1 * wdv + bdv;
    };

    // prefetch x(t=0)
    float xv = 0.0f;
    if (tid < BPW) xv = x[(size_t)(bbase + tid) * TSTEPS];

    for (int t = 0; t < TSTEPS; ++t) {
        __syncthreads();   // (a) h0 = h(t-1) ready; zbuf/xbuf consumed

        if (tid < BPW) {
            xbuf[tid] = xv * invn[tid];                       // fold l2-norm scale
            if (t + 1 < TSTEPS)                               // prefetch next x
                xv = x[(size_t)(bbase + tid) * TSTEPS + t + 1];
        }

        if (t > 0 && tid < BPW) layer1_step(t - 1);           // overlapped with WMMA

        // ---- layer-0 recurrent GEMM: z(t) = h(t-1) @ Wrec via WMMA ----
        // A (16-bit, 16x32): lane l: m = l&15; per K-half the 16 halves are the
        // two contiguous 8-half chunks k in [8*hi,+8) and [16+8*hi,+8).
        const _Float16* hrow = &h0[ln * UNITS];
        v8h a0lo = *(const v8h*)(hrow + 8 * hi);
        v8h a0hi = *(const v8h*)(hrow + 16 + 8 * hi);
        v8h a1lo = *(const v8h*)(hrow + 32 + 8 * hi);
        v8h a1hi = *(const v8h*)(hrow + 48 + 8 * hi);
        v16h A0 = __builtin_shufflevector(a0lo, a0hi, 0,1,2,3,4,5,6,7,8,9,10,11,12,13,14,15);
        v16h A1 = __builtin_shufflevector(a1lo, a1hi, 0,1,2,3,4,5,6,7,8,9,10,11,12,13,14,15);

        #pragma unroll
        for (int tt = 0; tt < 2; ++tt) {
            v8f acc = {};
            acc = __builtin_amdgcn_wmma_f32_16x16x32_f16(false, A0, false, Bfrag[tt][0],
                                                         (short)0, acc, false, false);
            acc = __builtin_amdgcn_wmma_f32_16x16x32_f16(false, A1, false, Bfrag[tt][1],
                                                         (short)0, acc, false, false);
            const int n = (2 * wave + tt) * 16 + ln;
            #pragma unroll
            for (int r = 0; r < 8; ++r)
                zbuf[(r + 8 * hi) * GATES + n] = acc[r];      // C/D layout: m=r+8*hi
        }

        __syncthreads();   // (b) zbuf(t), xbuf(t) ready; h0 reads done

        // ---- gate nonlinearities + cell update (i, j, f, o split) ----
        #pragma unroll
        for (int r2 = 0; r2 < 4; ++r2) {
            const int idx = tid + 256 * r2;
            const int m = idx >> 6, u = idx & 63;
            const float xt = xbuf[m];
            const float* zr = &zbuf[m * GATES];
            const float zi = zr[u]       + b0s[u]       + xt * w0r0[u];
            const float zj = zr[64 + u]  + b0s[64 + u]  + xt * w0r0[64 + u];
            const float zf = zr[128 + u] + b0s[128 + u] + xt * w0r0[128 + u];
            const float zo = zr[192 + u] + b0s[192 + u] + xt * w0r0[192 + u];
            float c = creg[r2];
            c = sigf(zf + 1.0f) * c + sigf(zi) * tanhfast(zj);  // FORGET_BIAS = 1.0
            creg[r2] = c;
            h0[m * UNITS + u] = (_Float16)(sigf(zo) * tanhfast(c));
        }
    }

    __syncthreads();
    if (tid < BPW) layer1_step(TSTEPS - 1);   // drain final layer-1/dense step
}

extern "C" void kernel_launch(void* const* d_in, const int* in_sizes, int n_in,
                              void* d_out, int out_size, void* d_ws, size_t ws_size,
                              hipStream_t stream) {
    (void)in_sizes; (void)n_in; (void)out_size; (void)d_ws; (void)ws_size;
    const float* x  = (const float*)d_in[0];
    const float* W0 = (const float*)d_in[1];
    const float* b0 = (const float*)d_in[2];
    const float* W1 = (const float*)d_in[3];
    const float* b1 = (const float*)d_in[4];
    const float* Wd = (const float*)d_in[5];
    const float* bd = (const float*)d_in[6];
    float* out = (float*)d_out;

    // 256 batches / 16 per workgroup = 16 workgroups of 8 wave32s
    lstm_fused_kernel<<<dim3(256 / BPW), dim3(256), 0, stream>>>(
        x, W0, b0, W1, b1, Wd, bd, out);
}
